// GroupedQueryAttention_34626026341103
// MI455X (gfx1250) — compile-verified, matrix operand order checked by reading, not run on hardware
//
#include <hip/hip_runtime.h>

#define B_   2
#define S_   2048
#define E_   2048
#define H_   32
#define KVH_ 8
#define D_   64

typedef __attribute__((ext_vector_type(16))) __bf16 v16bf;
typedef __attribute__((ext_vector_type(8)))  __bf16 v8bf;
typedef __attribute__((ext_vector_type(8)))  float  v8f;
typedef __attribute__((ext_vector_type(4)))  int    v4i;

// ---- gfx1250 async global->LDS copy (ASYNCcnt), guarded so either toolchain compiles
#if __has_builtin(__builtin_amdgcn_global_load_async_to_lds_b128) && \
    __has_builtin(__builtin_amdgcn_s_wait_asynccnt)
#define USE_ASYNC 1
#else
#define USE_ASYNC 0
#endif

#define GAS __attribute__((address_space(1)))
#define LAS __attribute__((address_space(3)))

#if USE_ASYNC
// 16 bytes per lane, per instruction; imm offset applies to BOTH addresses.
#define ASYNC_CP16(g, l, off) \
  __builtin_amdgcn_global_load_async_to_lds_b128((GAS v4i*)(g), (LAS v4i*)(l), (off), 0)
#define WAIT_ASYNC() __builtin_amdgcn_s_wait_asynccnt(0)
#else
#define WAIT_ASYNC() ((void)0)
#endif

static __device__ inline v8f vzero8() {
  v8f z;
#pragma unroll
  for (int i = 0; i < 8; ++i) z[i] = 0.0f;
  return z;
}

static __device__ inline v16bf combine16(const __bf16* lo, const __bf16* hi) {
  v8bf l = *(const v8bf*)lo;
  v8bf h = *(const v8bf*)hi;
  v16bf r;
#pragma unroll
  for (int i = 0; i < 8; ++i) { r[i] = l[i]; r[8 + i] = h[i]; }
  return r;
}

// Copy one 16-elem (32B) half-row: async if available, else temps (single wait).
static __device__ inline void cp_row32(const __bf16* g, __bf16* l) {
#if USE_ASYNC
  ASYNC_CP16(g, l, 0);
  ASYNC_CP16(g, l, 16);
#else
  v8bf x0 = *(const v8bf*)g;
  v8bf x1 = *(const v8bf*)(g + 8);
  *(v8bf*)l       = x0;
  *(v8bf*)(l + 8) = x1;
#endif
}

// ---------------------------------------------------------------------------
__global__ void cvt_f32_bf16(const float* __restrict__ in, __bf16* __restrict__ out, int n) {
  int i = blockIdx.x * blockDim.x + threadIdx.x;
  if (i < n) out[i] = (__bf16)in[i];
}

// ---------------------------------------------------------------------------
__global__ void rope_bf16(const __bf16* __restrict__ in, __bf16* __restrict__ out,
                          int heads, float qscale) {
  int i = blockIdx.x * blockDim.x + threadIdx.x;
  int cols = heads * D_;
  int total = B_ * S_ * cols;
  if (i >= total) return;
  int row = i / cols;
  int c = i - row * cols;
  int d = c & (D_ - 1);
  int s = row & (S_ - 1);
  int j = d & 31;
  float ang = (float)s * __powf(10000.0f, -(float)(2 * j) / (float)D_);
  float sn, cs;
  __sincosf(ang, &sn, &cs);
  float x = (float)in[i];
  float rot = (d < 32) ? -(float)in[i + 32] : (float)in[i - 32];
  out[i] = (__bf16)((x * cs + rot * sn) * qscale);
}

// ---------------------------------------------------------------------------
// NT GEMM: C[m,n] = sum_k A[m,k]*B[n,k].  WG tile 256x128, 8 waves (4Mx2N),
// wave tile 64x64 => 16 WMMAs per K-step on 8 loaded fragments.
// Double-buffered LDS (60KB), one barrier per K-step, async global->LDS.
// ---------------------------------------------------------------------------
#define GT_K 32
#define LSTR 40              // 32 elems + 8 pad; 80B rows, 16B-aligned chunks
#define LTA (256 * LSTR)
#define LTB (128 * LSTR)

__global__ __launch_bounds__(256)
void gemm_nt_bf16(const __bf16* __restrict__ A, const __bf16* __restrict__ Bm,
                  void* __restrict__ C, int M, int N, int K, int store_f32) {
  __shared__ __bf16 lA[2 * LTA];
  __shared__ __bf16 lB[2 * LTB];

  const int t    = threadIdx.x;
  const int lane = t & 31;
  const int wave = t >> 5;
  const int wm   = wave & 3;     // 4 M-waves * 64 rows
  const int wn   = wave >> 2;    // 2 N-waves * 64 cols
  const int m0   = blockIdx.y * 256;
  const int n0   = blockIdx.x * 128;
  const int lrow = lane & 15;
  const int ksel = lane >> 4;

  const int ldRow  = t >> 1;          // 0..127
  const int ldHalf = (t & 1) * 16;

  v8f acc[4][4];
#pragma unroll
  for (int i = 0; i < 4; ++i)
#pragma unroll
    for (int j = 0; j < 4; ++j) acc[i][j] = vzero8();

  const __bf16* aG0 = A  + (size_t)(m0 + ldRow) * K + ldHalf;
  const __bf16* aG1 = A  + (size_t)(m0 + 128 + ldRow) * K + ldHalf;
  const __bf16* bG  = Bm + (size_t)(n0 + ldRow) * K + ldHalf;
  const int lOffA0 = ldRow * LSTR + ldHalf;
  const int lOffA1 = (128 + ldRow) * LSTR + ldHalf;
  const int lOffB  = ldRow * LSTR + ldHalf;

  // prologue: fill buffer 0
  cp_row32(aG0, &lA[lOffA0]);
  cp_row32(aG1, &lA[lOffA1]);
  cp_row32(bG,  &lB[lOffB]);

  int cur = 0;
  for (int k0 = 0; k0 < K; k0 += GT_K) {
    WAIT_ASYNC();
    __syncthreads();

    if (k0 + GT_K < K) {
      int nb = cur ^ 1;
      cp_row32(aG0 + k0 + GT_K, &lA[nb * LTA + lOffA0]);
      cp_row32(aG1 + k0 + GT_K, &lA[nb * LTA + lOffA1]);
      cp_row32(bG  + k0 + GT_K, &lB[nb * LTB + lOffB]);
#if !USE_ASYNC
      __builtin_prefetch(aG0 + k0 + 2 * GT_K, 0, 1);
      __builtin_prefetch(aG1 + k0 + 2 * GT_K, 0, 1);
      __builtin_prefetch(bG  + k0 + 2 * GT_K, 0, 1);
#endif
    }

    const __bf16* cA = &lA[cur * LTA];
    const __bf16* cB = &lB[cur * LTB];
    v16bf afrag[4], bfrag[4];
#pragma unroll
    for (int mt = 0; mt < 4; ++mt) {
      const __bf16* p = &cA[(wm * 64 + mt * 16 + lrow) * LSTR + ksel * 8];
      afrag[mt] = combine16(p, p + 16);
    }
#pragma unroll
    for (int nt = 0; nt < 4; ++nt) {
      const __bf16* p = &cB[(wn * 64 + nt * 16 + lrow) * LSTR + ksel * 8];
      bfrag[nt] = combine16(p, p + 16);
    }
#pragma unroll
    for (int mt = 0; mt < 4; ++mt)
#pragma unroll
      for (int nt = 0; nt < 4; ++nt)
        acc[mt][nt] = __builtin_amdgcn_wmma_f32_16x16x32_bf16(
            false, afrag[mt], false, bfrag[nt], (short)0, acc[mt][nt], false, false);
    cur ^= 1;
  }

  const int mb = (lane >> 4) * 8;
#pragma unroll
  for (int mt = 0; mt < 4; ++mt)
#pragma unroll
    for (int nt = 0; nt < 4; ++nt) {
      int gn = n0 + wn * 64 + nt * 16 + lrow;
#pragma unroll
      for (int r = 0; r < 8; ++r) {
        int gm = m0 + wm * 64 + mt * 16 + mb + r;
        float v = acc[mt][nt][r];
        if (store_f32) ((float*)C)[(size_t)gm * N + gn] = v;
        else           ((__bf16*)C)[(size_t)gm * N + gn] = (__bf16)v;
      }
    }
}

// ---------------------------------------------------------------------------
// Flash attention: grid (S/128, H, B), 128 threads = 4 waves x 32 queries.
// K staged [key][d], V staged transposed [d][key]; V fragments reused across
// both query sub-tiles (2 M-frags per wave).
// ---------------------------------------------------------------------------
#define KSTR 72

__global__ __launch_bounds__(128)
void flash_attn(const __bf16* __restrict__ Q, const __bf16* __restrict__ Kc,
                const __bf16* __restrict__ Vc, __bf16* __restrict__ O) {
  __shared__ __bf16 lK [64 * KSTR];
  __shared__ __bf16 lVt[D_ * KSTR];
  __shared__ __bf16 lP [4 * 32 * KSTR];   // per-wave P [32][64]

  const int t    = threadIdx.x;
  const int lane = t & 31;
  const int wave = t >> 5;
  const int lrow = lane & 15;
  const int ksel = lane >> 4;
  const int qt = blockIdx.x, h = blockIdx.y, b = blockIdx.z;
  const int kvh = h >> 2;

  v16bf qf[2][2];
#pragma unroll
  for (int mt = 0; mt < 2; ++mt) {
    const int qrow = b * S_ + qt * 128 + wave * 32 + mt * 16 + lrow;
#pragma unroll
    for (int c = 0; c < 2; ++c) {
      const __bf16* p = Q + (size_t)qrow * (H_ * D_) + h * D_ + c * 32 + ksel * 8;
      qf[mt][c] = combine16(p, p + 16);
    }
  }

  float mi[2][8], li[2][8];
#pragma unroll
  for (int mt = 0; mt < 2; ++mt)
#pragma unroll
    for (int r = 0; r < 8; ++r) { mi[mt][r] = -3.0e38f; li[mt][r] = 0.0f; }
  v8f acc[2][4];
#pragma unroll
  for (int mt = 0; mt < 2; ++mt)
#pragma unroll
    for (int nt = 0; nt < 4; ++nt) acc[mt][nt] = vzero8();

  __bf16* myP = &lP[wave * 32 * KSTR];

  for (int kb = 0; kb < S_; kb += 64) {
    // ---- stage K (async) and V (transposed via temps) ----
#pragma unroll
    for (int it = 0; it < 2; ++it) {
      int c = t + it * 128;
      int key = c >> 2, seg = (c & 3) * 16;
      const __bf16* kg = Kc + (size_t)(b * S_ + kb + key) * (KVH_ * D_) + kvh * D_ + seg;
      cp_row32(kg, &lK[key * KSTR + seg]);
      const __bf16* vg = Vc + (size_t)(b * S_ + kb + key) * (KVH_ * D_) + kvh * D_ + seg;
      v8bf v0 = *(const v8bf*)vg;
      v8bf v1 = *(const v8bf*)(vg + 8);
#pragma unroll
      for (int i = 0; i < 8; ++i) {
        lVt[(seg + i)     * KSTR + key] = v0[i];
        lVt[(seg + 8 + i) * KSTR + key] = v1[i];
      }
    }
    WAIT_ASYNC();
    __syncthreads();

    // ---- scores: each K fragment pair feeds both query sub-tiles ----
    v8f sc[2][4];
#pragma unroll
    for (int nt = 0; nt < 4; ++nt) {
      const __bf16* p0 = &lK[(nt * 16 + lrow) * KSTR + 0 * 32 + ksel * 8];
      const __bf16* p1 = &lK[(nt * 16 + lrow) * KSTR + 1 * 32 + ksel * 8];
      v16bf kf0 = combine16(p0, p0 + 16);
      v16bf kf1 = combine16(p1, p1 + 16);
#pragma unroll
      for (int mt = 0; mt < 2; ++mt) {
        v8f z = vzero8();
        z = __builtin_amdgcn_wmma_f32_16x16x32_bf16(false, qf[mt][0], false, kf0, (short)0, z, false, false);
        z = __builtin_amdgcn_wmma_f32_16x16x32_bf16(false, qf[mt][1], false, kf1, (short)0, z, false, false);
        sc[mt][nt] = z;
      }
    }

    // ---- online softmax per query sub-tile ----
    const int mbs = ksel * 8;
#pragma unroll
    for (int mt = 0; mt < 2; ++mt) {
      float scalev[8];
#pragma unroll
      for (int r = 0; r < 8; ++r) {
        float m = fmaxf(fmaxf(sc[mt][0][r], sc[mt][1][r]), fmaxf(sc[mt][2][r], sc[mt][3][r]));
#pragma unroll
        for (int mask = 1; mask < 16; mask <<= 1)
          m = fmaxf(m, __shfl_xor(m, mask, 32));
        float mn = fmaxf(mi[mt][r], m);
        scalev[r] = __expf(mi[mt][r] - mn);
        mi[mt][r] = mn;
      }
#pragma unroll
      for (int nt = 0; nt < 4; ++nt)
#pragma unroll
        for (int r = 0; r < 8; ++r)
          sc[mt][nt][r] = __expf(sc[mt][nt][r] - mi[mt][r]);
#pragma unroll
      for (int r = 0; r < 8; ++r) {
        float s = sc[mt][0][r] + sc[mt][1][r] + sc[mt][2][r] + sc[mt][3][r];
#pragma unroll
        for (int mask = 1; mask < 16; mask <<= 1)
          s += __shfl_xor(s, mask, 32);
        li[mt][r] = li[mt][r] * scalev[r] + s;
#pragma unroll
        for (int nt = 0; nt < 4; ++nt) acc[mt][nt][r] *= scalev[r];
      }
      // C-layout -> A-layout bounce through per-wave LDS scratch
#pragma unroll
      for (int nt = 0; nt < 4; ++nt)
#pragma unroll
        for (int r = 0; r < 8; ++r)
          myP[(mt * 16 + mbs + r) * KSTR + nt * 16 + lrow] = (__bf16)sc[mt][nt][r];
    }

    v16bf pf[2][2];
#pragma unroll
    for (int mt = 0; mt < 2; ++mt)
#pragma unroll
      for (int c = 0; c < 2; ++c) {
        const __bf16* p = &myP[(mt * 16 + lrow) * KSTR + c * 32 + ksel * 8];
        pf[mt][c] = combine16(p, p + 16);
      }
    // ---- O += P*V : each V fragment reused by both query sub-tiles ----
#pragma unroll
    for (int nt = 0; nt < 4; ++nt)
#pragma unroll
      for (int c = 0; c < 2; ++c) {
        const __bf16* p = &lVt[(nt * 16 + lrow) * KSTR + c * 32 + ksel * 8];
        v16bf vf = combine16(p, p + 16);
#pragma unroll
        for (int mt = 0; mt < 2; ++mt)
          acc[mt][nt] = __builtin_amdgcn_wmma_f32_16x16x32_bf16(
              false, pf[mt][c], false, vf, (short)0, acc[mt][nt], false, false);
      }
    __syncthreads();
  }

  const int mbs = ksel * 8;
#pragma unroll
  for (int mt = 0; mt < 2; ++mt)
#pragma unroll
    for (int nt = 0; nt < 4; ++nt)
#pragma unroll
      for (int r = 0; r < 8; ++r) {
        int q = qt * 128 + wave * 32 + mt * 16 + mbs + r;
        float v = acc[mt][nt][r] / li[mt][r];
        O[(size_t)(b * S_ + q) * E_ + h * D_ + nt * 16 + lrow] = (__bf16)v;
      }
}

// ---------------------------------------------------------------------------
extern "C" void kernel_launch(void* const* d_in, const int* in_sizes, int n_in,
                              void* d_out, int out_size, void* d_ws, size_t ws_size,
                              hipStream_t stream) {
  const float* x  = (const float*)d_in[0];
  const float* Wq = (const float*)d_in[1];
  const float* Wk = (const float*)d_in[2];
  const float* Wv = (const float*)d_in[3];
  const float* Wo = (const float*)d_in[4];

  const int ROWS = B_ * S_;
  const int KVE  = KVH_ * D_;
  const size_t nX  = (size_t)ROWS * E_;
  const size_t nWq = (size_t)E_ * E_;
  const size_t nWk = (size_t)KVE * E_;
  const size_t nKV = (size_t)ROWS * KVE;

  char* ws = (char*)d_ws;
  size_t off = 0;
  auto alloc = [&](size_t bytes) -> char* {
    char* p = ws + off;
    off += (bytes + 255) & ~(size_t)255;
    return p;
  };
  __bf16* xb  = (__bf16*)alloc(nX  * 2);
  __bf16* Wqb = (__bf16*)alloc(nWq * 2);
  __bf16* Wkb = (__bf16*)alloc(nWk * 2);
  __bf16* Wvb = (__bf16*)alloc(nWk * 2);
  __bf16* Wob = (__bf16*)alloc(nWq * 2);
  __bf16* Qp  = (__bf16*)alloc(nX  * 2);
  __bf16* Kp  = (__bf16*)alloc(nKV * 2);
  __bf16* Vb  = (__bf16*)alloc(nKV * 2);
  __bf16* Qr  = (__bf16*)alloc(nX  * 2);
  __bf16* Kr  = (__bf16*)alloc(nKV * 2);
  __bf16* Ao  = (__bf16*)alloc(nX  * 2);

  auto cvt = [&](const float* src, __bf16* dst, size_t n) {
    cvt_f32_bf16<<<(unsigned)((n + 255) / 256), 256, 0, stream>>>(src, dst, (int)n);
  };
  cvt(x,  xb,  nX);
  cvt(Wq, Wqb, nWq);
  cvt(Wk, Wkb, nWk);
  cvt(Wv, Wvb, nWk);
  cvt(Wo, Wob, nWq);

  dim3 blk(256);
  gemm_nt_bf16<<<dim3(E_ / 128,  ROWS / 256), blk, 0, stream>>>(xb, Wqb, Qp, ROWS, E_,  E_, 0);
  gemm_nt_bf16<<<dim3(KVE / 128, ROWS / 256), blk, 0, stream>>>(xb, Wkb, Kp, ROWS, KVE, E_, 0);
  gemm_nt_bf16<<<dim3(KVE / 128, ROWS / 256), blk, 0, stream>>>(xb, Wvb, Vb, ROWS, KVE, E_, 0);

  rope_bf16<<<(unsigned)((nX  + 255) / 256), 256, 0, stream>>>(Qp, Qr, H_,   0.125f);
  rope_bf16<<<(unsigned)((nKV + 255) / 256), 256, 0, stream>>>(Kp, Kr, KVH_, 1.0f);

  flash_attn<<<dim3(S_ / 128, H_, B_), dim3(128), 0, stream>>>(Qr, Kr, Vb, Ao);

  gemm_nt_bf16<<<dim3(E_ / 128, ROWS / 256), blk, 0, stream>>>(Ao, Wob, d_out, ROWS, E_, E_, 1);
}
